// UncertaintySemiMarkovCRFHead_1022202216822
// MI455X (gfx1250) — matrix operand order repaired and checked by reading, not compiled, not run on hardware
//
#include <hip/hip_runtime.h>
#include <math.h>

typedef __attribute__((ext_vector_type(16))) __bf16 v16bf;
typedef __attribute__((ext_vector_type(8)))  float  v8f;

#define NEGF (-1e30f)

__device__ __forceinline__ void lse_acc(float& mx, float& sm, float v) {
  float nm = fmaxf(mx, v);
  sm = sm * __expf(mx - nm) + __expf(v - nm);
  mx = nm;
}
__device__ __forceinline__ void lse_merge(float& mx, float& sm, float mx2, float sm2) {
  float nm = fmaxf(mx, mx2);
  sm = sm * __expf(mx - nm) + sm2 * __expf(mx2 - nm);
  mx = nm;
}
__device__ __forceinline__ __bf16 f2bf(float x) { return (__bf16)x; }

// ---------------------------------------------------------------------------
// Kernel 1: scores = hidden @ W + b  via v_wmma_f32_16x16x32_bf16
// hidden: (rows, H) fp32 row-major; W: (H, C) fp32; scores: (rows, C) fp32.
// One wave per 16-row M tile; N padded to 32 (two 16-wide WMMA tiles).
// W chunk staged in LDS transposed to [n][k] bf16 so B-fragments are
// contiguous 32B ds reads per lane.
// ---------------------------------------------------------------------------
#define PROJ_KB 256

__global__ __launch_bounds__(256) void proj_wmma_kernel(
    const float* __restrict__ hidden, const float* __restrict__ W,
    const float* __restrict__ bias, float* __restrict__ scores,
    int H, int C, int rows)
{
  __shared__ __align__(32) __bf16 wt[32][PROJ_KB];
  const int lane = threadIdx.x & 31;
  const int wave = threadIdx.x >> 5;
  const int tile = blockIdx.x * 8 + wave;
  const int row0 = tile * 16;
  const int g  = lane >> 4;     // half-wave group
  const int mr = lane & 15;     // M row within tile (A) / N col (B,C)
  int arow = row0 + mr; if (arow >= rows) arow = rows - 1;
  const float* hrow = hidden + (size_t)arow * H;

  v8f acc0 = {0.f,0.f,0.f,0.f,0.f,0.f,0.f,0.f};
  v8f acc1 = {0.f,0.f,0.f,0.f,0.f,0.f,0.f,0.f};

  for (int kb = 0; kb < H; kb += PROJ_KB) {
    // cooperative W chunk load, transposed + bf16-converted, zero-padded N
    for (int idx = threadIdx.x; idx < PROJ_KB * 32; idx += blockDim.x) {
      int kk = idx >> 5, n = idx & 31;
      float v = (n < C && (kb + kk) < H) ? W[(size_t)(kb + kk) * C + n] : 0.f;
      wt[n][kk] = f2bf(v);
    }
    __syncthreads();
    for (int ks = 0; ks < PROJ_KB && (kb + ks) < H; ks += 32) {
      const float* hp = hrow + kb + ks;
      // A fragment: lane holds K = {g*16..g*16+15}
      float4 f0 = *(const float4*)(hp + g * 16);
      float4 f1 = *(const float4*)(hp + g * 16 + 4);
      float4 f2 = *(const float4*)(hp + g * 16 + 8);
      float4 f3 = *(const float4*)(hp + g * 16 + 12);
      v16bf a;
      a[0]=f2bf(f0.x);  a[1]=f2bf(f0.y);  a[2]=f2bf(f0.z);  a[3]=f2bf(f0.w);
      a[4]=f2bf(f1.x);  a[5]=f2bf(f1.y);  a[6]=f2bf(f1.z);  a[7]=f2bf(f1.w);
      a[8]=f2bf(f2.x);  a[9]=f2bf(f2.y);  a[10]=f2bf(f2.z); a[11]=f2bf(f2.w);
      a[12]=f2bf(f3.x); a[13]=f2bf(f3.y); a[14]=f2bf(f3.z); a[15]=f2bf(f3.w);
      // B fragments: lane = column N, K = ks + g*16 + e (contiguous in LDS)
      int koff = ks + g * 16;
      v16bf b0 = *(const v16bf*)(&wt[mr][koff]);
      v16bf b1 = *(const v16bf*)(&wt[mr + 16][koff]);
      acc0 = __builtin_amdgcn_wmma_f32_16x16x32_bf16(false, a, false, b0, (short)0, acc0, false, false);
      acc1 = __builtin_amdgcn_wmma_f32_16x16x32_bf16(false, a, false, b1, (short)0, acc1, false, false);
    }
    __syncthreads();
  }
  // C layout: VGPR j -> M = j + 8*g, N = lane&15 (+16 for acc1)
  #pragma unroll
  for (int j = 0; j < 8; ++j) {
    int rowm = row0 + j + 8 * g;
    if (rowm < rows) {
      int c0 = mr;
      if (c0 < C) scores[(size_t)rowm * C + c0] = acc0[j] + bias[c0];
      int c1 = 16 + mr;
      if (c1 < C) scores[(size_t)rowm * C + c1] = acc1[j] + bias[c1];
    }
  }
}

// ---------------------------------------------------------------------------
// Kernel 2: cum[b, t+1, c] = prefix-sum over t of scores[b, t, c]; cum[b,0,c]=0
// One block per (b, c) sequence; 256-wide Hillis-Steele scan with carry.
// ---------------------------------------------------------------------------
__global__ __launch_bounds__(256) void cumsum_kernel(
    const float* __restrict__ scores, float* __restrict__ cum, int T, int C)
{
  __shared__ float buf[256];
  int b = blockIdx.x / C;
  int c = blockIdx.x % C;
  int tid = threadIdx.x;
  const size_t sbase = (size_t)b * T * C;
  const size_t cbase = (size_t)b * (T + 1) * C;
  if (tid == 0) cum[cbase + c] = 0.f;
  float carry = 0.f;
  for (int t0 = 0; t0 < T; t0 += 256) {
    int t = t0 + tid;
    float x = (t < T) ? scores[sbase + (size_t)t * C + c] : 0.f;
    buf[tid] = x;
    __syncthreads();
    for (int off = 1; off < 256; off <<= 1) {
      float y = (tid >= off) ? buf[tid - off] : 0.f;
      __syncthreads();
      buf[tid] += y;
      __syncthreads();
    }
    if (t < T) cum[cbase + (size_t)(t + 1) * C + c] = carry + buf[tid];
    carry += buf[255];
    __syncthreads();
  }
}

// ---------------------------------------------------------------------------
// Kernel 3: forward + backward semi-CRF DP. One wave32 per block, lane = class.
// Blocks [0,B): forward (alpha, m, logZ). Blocks [B,2B): backward (beta, s).
// Ring buffers + small tables in LDS (dynamic indices stay off scratch).
// ---------------------------------------------------------------------------
__global__ __launch_bounds__(32) void semicrf_dp_kernel(
    const float* __restrict__ cum, const float* __restrict__ trans,
    const float* __restrict__ dbias, const int* __restrict__ lengths,
    float* __restrict__ alpha_g, float* __restrict__ m_g,
    float* __restrict__ beta_g, float* __restrict__ s_g,
    float* __restrict__ logZ_g, int B, int T, int C, int K)
{
  const int lane = threadIdx.x;
  const bool fwd = ((int)blockIdx.x < B);
  const int b = fwd ? (int)blockIdx.x : ((int)blockIdx.x - B);
  const int c = lane;
  const bool act = (c < C);
  const int Km = K - 1;
  const int L = lengths[b];

  __shared__ float ringA[32][33];   // fwd: m history   | bwd: h = beta + cum
  __shared__ float ringB[32][33];   // fwd: cum history
  __shared__ float dbsh[32][33];    // duration_bias[k][c] per lane column
  __shared__ float trsh[32][33];    // fwd: trans[c'][c] | bwd: trans[c][c'']
  __shared__ float sh[33];

  #pragma unroll
  for (int k = 0; k < 32; ++k)
    dbsh[k][lane] = (k < K && act) ? dbias[(size_t)k * C + c] : NEGF;

  const size_t base = (size_t)b * (T + 1) * C;
  const float* cumb = cum + base;

  if (fwd) {
    #pragma unroll
    for (int cp = 0; cp < 32; ++cp)
      trsh[cp][lane] = (cp < C && act) ? trans[(size_t)cp * C + c] : NEGF;
    #pragma unroll
    for (int t = 0; t < 32; ++t) { ringA[t][lane] = NEGF; ringB[t][lane] = 0.f; }
    __syncthreads();
    // m[0,c] = lse_{c'} (alpha0=0 + trans[c',c])
    float mx = NEGF, sm = 0.f;
    for (int cp = 0; cp < C; ++cp) lse_acc(mx, sm, trsh[cp][lane]);
    float m0 = mx + __logf(sm);
    ringA[0][lane] = m0;
    ringB[0][lane] = act ? cumb[c] : 0.f;
    if (act) { alpha_g[base + c] = 0.f; m_g[base + c] = m0; }
    float alphaL = NEGF;
    __syncthreads();
    for (int t = 1; t <= T; ++t) {
      float cum_t = act ? cumb[(size_t)t * C + c] : 0.f;
      float mx0=NEGF,sm0=0.f,mx1=NEGF,sm1=0.f,mx2=NEGF,sm2=0.f,mx3=NEGF,sm3=0.f;
      int k = 1;
      for (; k + 3 <= Km; k += 4) {
        int i0=(t-k)&31, i1=(t-k-1)&31, i2=(t-k-2)&31, i3=(t-k-3)&31;
        lse_acc(mx0,sm0, ringA[i0][lane]-ringB[i0][lane]+dbsh[k  ][lane]);
        lse_acc(mx1,sm1, ringA[i1][lane]-ringB[i1][lane]+dbsh[k+1][lane]);
        lse_acc(mx2,sm2, ringA[i2][lane]-ringB[i2][lane]+dbsh[k+2][lane]);
        lse_acc(mx3,sm3, ringA[i3][lane]-ringB[i3][lane]+dbsh[k+3][lane]);
      }
      for (; k <= Km; ++k) {
        int i0=(t-k)&31;
        lse_acc(mx0,sm0, ringA[i0][lane]-ringB[i0][lane]+dbsh[k][lane]);
      }
      lse_merge(mx0,sm0,mx1,sm1);
      lse_merge(mx2,sm2,mx3,sm3);
      lse_merge(mx0,sm0,mx2,sm2);
      float alpha_t = cum_t + mx0 + __logf(sm0);
      sh[lane] = act ? alpha_t : NEGF;
      __syncthreads();
      float mxm = NEGF, smm = 0.f;
      for (int cp = 0; cp < C; ++cp) lse_acc(mxm, smm, sh[cp] + trsh[cp][lane]);
      float m_t = mxm + __logf(smm);
      __syncthreads();
      ringA[t & 31][lane] = m_t;
      ringB[t & 31][lane] = cum_t;
      if (act) {
        alpha_g[base + (size_t)t * C + c] = alpha_t;
        m_g[base + (size_t)t * C + c] = m_t;
      }
      if (t == L) alphaL = alpha_t;
      __syncthreads();
    }
    sh[lane] = act ? alphaL : NEGF;
    __syncthreads();
    if (lane == 0) {
      float mz = NEGF, sz = 0.f;
      for (int cp = 0; cp < C; ++cp) lse_acc(mz, sz, sh[cp]);
      logZ_g[b] = mz + __logf(sz);
    }
  } else {
    #pragma unroll
    for (int cd = 0; cd < 32; ++cd)
      trsh[cd][lane] = (cd < C && act) ? trans[(size_t)c * C + cd] : NEGF;
    #pragma unroll
    for (int t = 0; t < 32; ++t) ringA[t][lane] = NEGF;
    __syncthreads();
    float cumL = act ? cumb[(size_t)L * C + c] : 0.f;
    ringA[L & 31][lane] = cumL;                  // h[L] = beta[L](=0) + cum[L]
    if (act) beta_g[base + (size_t)L * C + c] = 0.f;
    __syncthreads();
    for (int t = L - 1; t >= 0; --t) {
      float cum_t = act ? cumb[(size_t)t * C + c] : 0.f;
      // s[t,c] = lse_k ( db[k,c] + h[t+k,c] )
      float mx0=NEGF,sm0=0.f,mx1=NEGF,sm1=0.f,mx2=NEGF,sm2=0.f,mx3=NEGF,sm3=0.f;
      int k = 1;
      for (; k + 3 <= Km; k += 4) {
        int i0=(t+k)&31, i1=(t+k+1)&31, i2=(t+k+2)&31, i3=(t+k+3)&31;
        lse_acc(mx0,sm0, ringA[i0][lane]+dbsh[k  ][lane]);
        lse_acc(mx1,sm1, ringA[i1][lane]+dbsh[k+1][lane]);
        lse_acc(mx2,sm2, ringA[i2][lane]+dbsh[k+2][lane]);
        lse_acc(mx3,sm3, ringA[i3][lane]+dbsh[k+3][lane]);
      }
      for (; k <= Km; ++k) {
        int i0=(t+k)&31;
        lse_acc(mx0,sm0, ringA[i0][lane]+dbsh[k][lane]);
      }
      lse_merge(mx0,sm0,mx1,sm1);
      lse_merge(mx2,sm2,mx3,sm3);
      lse_merge(mx0,sm0,mx2,sm2);
      float s_t = mx0 + __logf(sm0);
      sh[lane] = act ? (s_t - cum_t) : NEGF;
      __syncthreads();
      // beta[t,c'] = lse_c ( trans[c',c] + s[t,c] - cum[t,c] )
      float mxb = NEGF, smb = 0.f;
      for (int cd = 0; cd < C; ++cd) lse_acc(mxb, smb, trsh[cd][lane] + sh[cd]);
      float beta_t = mxb + __logf(smb);
      __syncthreads();
      ringA[t & 31][lane] = beta_t + cum_t;
      if (act) {
        beta_g[base + (size_t)t * C + c] = beta_t;
        s_g[base + (size_t)t * C + c] = s_t;
      }
      __syncthreads();
    }
  }
}

// ---------------------------------------------------------------------------
// Kernel 4: boundary_raw[b,u-1] = sum_c | exp(a+b-lz) - exp(m-cum+s-lz) |
// ---------------------------------------------------------------------------
__global__ __launch_bounds__(256) void marginal_kernel(
    const float* __restrict__ alpha, const float* __restrict__ beta,
    const float* __restrict__ m, const float* __restrict__ s,
    const float* __restrict__ cum, const float* __restrict__ logZ,
    const int* __restrict__ lengths, float* __restrict__ raw,
    int B, int T, int C)
{
  int gid = blockIdx.x * blockDim.x + threadIdx.x;
  if (gid >= B * T) return;
  int b = gid / T;
  int u = gid % T + 1;
  int L = lengths[b];
  float lz = logZ[b];
  const size_t base = (size_t)b * (T + 1) * C + (size_t)u * C;
  float acc = 0.f;
  if (u <= L) {
    for (int c = 0; c < C; ++c) {
      float ee = __expf(alpha[base + c] + beta[base + c] - lz);
      float es = (u < L) ? __expf(m[base + c] - cum[base + c] + s[base + c] - lz) : 0.f;
      acc += fabsf(ee - es);
    }
  }
  raw[gid] = acc;
}

// ---------------------------------------------------------------------------
// Kernel 5: per-batch normalize by (max + 1e-8)
// ---------------------------------------------------------------------------
__global__ __launch_bounds__(256) void normalize_kernel(
    const float* __restrict__ raw, float* __restrict__ out, int T)
{
  __shared__ float red[256];
  int b = blockIdx.x, tid = threadIdx.x;
  const size_t base = (size_t)b * T;
  float mx = 0.f;
  for (int t = tid; t < T; t += 256) mx = fmaxf(mx, raw[base + t]);
  red[tid] = mx;
  __syncthreads();
  for (int off = 128; off > 0; off >>= 1) {
    if (tid < off) red[tid] = fmaxf(red[tid], red[tid + off]);
    __syncthreads();
  }
  float inv = 1.f / (red[0] + 1e-8f);
  for (int t = tid; t < T; t += 256) out[base + t] = raw[base + t] * inv;
}

// ---------------------------------------------------------------------------
extern "C" void kernel_launch(void* const* d_in, const int* in_sizes, int n_in,
                              void* d_out, int out_size, void* d_ws, size_t ws_size,
                              hipStream_t stream) {
  const float* hidden = (const float*)d_in[0];
  const int*   lengths = (const int*)d_in[1];
  const float* projw  = (const float*)d_in[2];
  const float* projb  = (const float*)d_in[3];
  const float* trans  = (const float*)d_in[4];
  const float* dbias  = (const float*)d_in[5];
  float* out = (float*)d_out;

  int B = in_sizes[1];
  int C = (int)(sqrt((double)in_sizes[4]) + 0.5);
  int H = in_sizes[2] / C;
  int T = in_sizes[0] / (B * H);
  int K = in_sizes[5] / C;

  float* w = (float*)d_ws;
  size_t Ssc = (size_t)B * T * C;
  size_t Scm = (size_t)B * (T + 1) * C;
  float* scores = w;
  float* cum    = scores + Ssc;
  float* alpha  = cum + Scm;
  float* mArr   = alpha + Scm;
  float* beta   = mArr + Scm;
  float* sArr   = beta + Scm;
  float* raw    = sArr + Scm;
  float* logZ   = raw + (size_t)B * T;

  int rows = B * T;
  int tiles = (rows + 15) / 16;
  int pblocks = (tiles + 7) / 8;
  proj_wmma_kernel<<<pblocks, 256, 0, stream>>>(hidden, projw, projb, scores, H, C, rows);
  cumsum_kernel<<<B * C, 256, 0, stream>>>(scores, cum, T, C);
  semicrf_dp_kernel<<<2 * B, 32, 0, stream>>>(cum, trans, dbias, lengths,
                                              alpha, mArr, beta, sArr, logZ,
                                              B, T, C, K);
  int total = B * T;
  marginal_kernel<<<(total + 255) / 256, 256, 0, stream>>>(alpha, beta, mArr, sArr,
                                                           cum, logZ, lengths, raw,
                                                           B, T, C);
  normalize_kernel<<<B, 256, 0, stream>>>(raw, out, T);
}
